// SelfAttention_18897856102400
// MI455X (gfx1250) — matrix operand order repaired and checked by reading, not compiled, not run on hardware
//
#include <hip/hip_runtime.h>
#include <hip/hip_bf16.h>

// ---------------------------------------------------------------------------
// Self-attention for MI455X (gfx1250, wave32, WMMA f32_16x16x32_f16)
//   x:[4096,1024] f32,  Wq/Wk/Wv:[1024,1024] f32  ->  out:[4096,1024] f32
// 4 passes:
//   1) Q = (x Wq)/sqrt(d), K = x Wk, Vt = (x Wv)^T   (f16 in workspace)
//   2) S = Q K^T  (NT WMMA GEMM, f16 out, 32MB - L2 resident)
//   3) row softmax of S in place (f32 math)
//   4) out = P Vt^T (NT WMMA GEMM, f32 out)
// ---------------------------------------------------------------------------

#define SEQ    4096
#define DMODEL 1024

typedef _Float16 v16h __attribute__((ext_vector_type(16)));
typedef _Float16 v8h  __attribute__((ext_vector_type(8)));
typedef float    v8f  __attribute__((ext_vector_type(8)));

// --- WMMA fragment helpers (layouts per CDNA5 ISA 7.12.2, wave32) -----------

// A-matrix 16x32 f16: lane L holds row M=L%16; halfs 0..7 = K[(L/16)*8 .. +7],
// halfs 8..15 = K[16+(L/16)*8 .. +7].
__device__ __forceinline__ v16h load_a_frag(const _Float16* rowptr, int half) {
    v8h lo = *(const v8h*)(rowptr + half * 8);
    v8h hi = *(const v8h*)(rowptr + 16 + half * 8);
    return __builtin_shufflevector(lo, hi, 0,1,2,3,4,5,6,7,8,9,10,11,12,13,14,15);
}

// B-matrix 32x16 f16: lane L holds col N=L%16; halfs 0..7 = K[(L/16)*8 .. +7],
// halfs 8..15 = K[16+(L/16)*8 .. +7].
// rowptr points at the K-contiguous row for column N (NT layout).
__device__ __forceinline__ v16h load_b_frag(const _Float16* rowptr, int half) {
    v8h lo = *(const v8h*)(rowptr + half * 8);
    v8h hi = *(const v8h*)(rowptr + 16 + half * 8);
    return __builtin_shufflevector(lo, hi, 0,1,2,3,4,5,6,7,8,9,10,11,12,13,14,15);
}

__device__ __forceinline__ v8f wmma_f16(v16h a, v16h b, v8f c) {
    // (neg_a, A, neg_b, B, c_mod, C, reuse_a, reuse_b)
    return __builtin_amdgcn_wmma_f32_16x16x32_f16(false, a, false, b, (short)0, c,
                                                  false, false);
}

// ---------------------------------------------------------------------------
// Pass 1: one projection  out = x @ W  (scaled; optionally stored transposed)
//   x:[SEQ,DMODEL] f32 row-major, W:[DMODEL,DMODEL] f32 row-major
//   transpose_out==0: out[m*DMODEL+n] = f16(acc*scale)
//   transpose_out==1: out[n*SEQ+m]    = f16(acc*scale)   (for V^T)
// grid: (SEQ/64, DMODEL/64), block: 256 threads (8 wave32)
// ---------------------------------------------------------------------------
#define PBM 64
#define PBN 64
#define PBK 32
#define PLDST (PBK + 8)          // 40 halfs = 80B row stride (16B aligned)

__global__ __launch_bounds__(256)
void qkv_project_kernel(const float* __restrict__ x, const float* __restrict__ W,
                        _Float16* __restrict__ out, float scale, int transpose_out) {
    __shared__ __align__(16) _Float16 As[PBM][PLDST];  // x tile, row-major (m,k)
    __shared__ __align__(16) _Float16 Ws[PBN][PLDST];  // W tile, transposed (n,k)

    const int t    = threadIdx.x;
    const int wave = t >> 5;
    const int lane = t & 31;
    const int rr   = lane & 15;
    const int half = lane >> 4;
    const int tm   = wave & 3;          // 16-row tile within 64
    const int tn0  = (wave >> 2) * 2;   // two 16-col tiles within 64

    const int row0 = blockIdx.x * PBM;
    const int col0 = blockIdx.y * PBN;

    v8f acc0 = {}; v8f acc1 = {};

    for (int k0 = 0; k0 < DMODEL; k0 += PBK) {
        // stage x[64x32] -> As (f32->f16), coalesced 32B per thread
        {
            const int r = t >> 2;
            const int c = (t & 3) * 8;
            const float4* src = (const float4*)(x + (size_t)(row0 + r) * DMODEL + k0 + c);
            float4 f0 = src[0], f1 = src[1];
            As[r][c+0] = (_Float16)f0.x; As[r][c+1] = (_Float16)f0.y;
            As[r][c+2] = (_Float16)f0.z; As[r][c+3] = (_Float16)f0.w;
            As[r][c+4] = (_Float16)f1.x; As[r][c+5] = (_Float16)f1.y;
            As[r][c+6] = (_Float16)f1.z; As[r][c+7] = (_Float16)f1.w;
        }
        // stage W[32x64] -> Ws transposed (n,k), coalesced global reads
        {
            const int kr = t >> 3;
            const int nc = (t & 7) * 8;
            const float4* src = (const float4*)(W + (size_t)(k0 + kr) * DMODEL + col0 + nc);
            float4 w0 = src[0], w1 = src[1];
            Ws[nc+0][kr] = (_Float16)w0.x; Ws[nc+1][kr] = (_Float16)w0.y;
            Ws[nc+2][kr] = (_Float16)w0.z; Ws[nc+3][kr] = (_Float16)w0.w;
            Ws[nc+4][kr] = (_Float16)w1.x; Ws[nc+5][kr] = (_Float16)w1.y;
            Ws[nc+6][kr] = (_Float16)w1.z; Ws[nc+7][kr] = (_Float16)w1.w;
        }
        __syncthreads();

        v16h a  = load_a_frag(&As[tm * 16 + rr][0], half);
        v16h b0 = load_b_frag(&Ws[tn0 * 16 + rr][0], half);
        v16h b1 = load_b_frag(&Ws[(tn0 + 1) * 16 + rr][0], half);
        acc0 = wmma_f16(a, b0, acc0);
        acc1 = wmma_f16(a, b1, acc1);
        __syncthreads();
    }

    // store (C layout: VGPR i -> row tm*16 + half*8 + i, col tn*16 + rr)
    const int mbase = row0 + tm * 16 + half * 8;
    for (int which = 0; which < 2; ++which) {
        v8f acc = which ? acc1 : acc0;
        const int n = col0 + (tn0 + which) * 16 + rr;
        if (transpose_out) {
            v8h pk;
            #pragma unroll
            for (int i = 0; i < 8; ++i) pk[i] = (_Float16)(acc[i] * scale);
            *(v8h*)(out + (size_t)n * SEQ + mbase) = pk;     // contiguous in m
        } else {
            #pragma unroll
            for (int i = 0; i < 8; ++i)
                out[(size_t)(mbase + i) * DMODEL + n] = (_Float16)(acc[i] * scale);
        }
    }
}

// ---------------------------------------------------------------------------
// Passes 2 & 4: NT GEMM  C[m,n] = sum_k A[m,k]*B[n,k]   (A,B f16 row-major)
// Block tile 128x64, BK=64, 8 wave32s each owning a 32x32 output tile
// (4 WMMA accumulators), double-buffered LDS with global->reg prefetch so
// global_load latency overlaps v_wmma issue; one barrier per K-tile.
// grid: (M/128, N/64), block: 256 threads
// ---------------------------------------------------------------------------
#define GBM 128
#define GBN 64
#define GBK 64
#define GLDST (GBK + 8)          // 72 halfs = 144B row stride (16B aligned)

template <int OUT_F16>
__global__ __launch_bounds__(256)
void gemm_nt_kernel(const _Float16* __restrict__ A, const _Float16* __restrict__ B,
                    void* __restrict__ C, int K, int lda, int ldb, int ldc) {
    __shared__ __align__(16) _Float16 As[2][GBM][GLDST];   // 36 KB
    __shared__ __align__(16) _Float16 Bs[2][GBN][GLDST];   // 18 KB

    const int t    = threadIdx.x;
    const int wave = t >> 5;
    const int lane = t & 31;
    const int rr   = lane & 15;
    const int half = lane >> 4;
    const int wr   = wave & 3;        // 32-row strip within 128
    const int wc   = wave >> 2;       // 32-col strip within 64

    const int row0 = blockIdx.x * GBM;
    const int col0 = blockIdx.y * GBN;

    // staging maps: A 128x64 halfs -> 32 halfs/thread; B 64x64 -> 16/thread
    const int ar = t >> 1;            // 0..127
    const int ac = (t & 1) * 32;      // 0 | 32
    const int br = t >> 2;            // 0..63
    const int bc = (t & 3) * 16;      // 0,16,32,48

    const _Float16* agp = A + (size_t)(row0 + ar) * lda + ac;
    const _Float16* bgp = B + (size_t)(col0 + br) * ldb + bc;

    v8h areg[4], breg[2];

    // prologue: K-tile 0 -> buffer 0
    #pragma unroll
    for (int i = 0; i < 4; ++i) areg[i] = *(const v8h*)(agp + i * 8);
    #pragma unroll
    for (int i = 0; i < 2; ++i) breg[i] = *(const v8h*)(bgp + i * 8);
    #pragma unroll
    for (int i = 0; i < 4; ++i) *(v8h*)&As[0][ar][ac + i * 8] = areg[i];
    #pragma unroll
    for (int i = 0; i < 2; ++i) *(v8h*)&Bs[0][br][bc + i * 8] = breg[i];
    __syncthreads();

    v8f acc00 = {}, acc01 = {}, acc10 = {}, acc11 = {};

    const int ktiles = K / GBK;
    for (int kt = 0; kt < ktiles; ++kt) {
        const int  cur  = kt & 1;
        const bool more = (kt + 1) < ktiles;

        // issue next K-tile's global loads before the math (latency overlap)
        if (more) {
            const _Float16* ag = agp + (size_t)(kt + 1) * GBK;
            const _Float16* bg = bgp + (size_t)(kt + 1) * GBK;
            #pragma unroll
            for (int i = 0; i < 4; ++i) areg[i] = *(const v8h*)(ag + i * 8);
            #pragma unroll
            for (int i = 0; i < 2; ++i) breg[i] = *(const v8h*)(bg + i * 8);
        }

        // 8 WMMAs on the current buffer (A/B fragments each reused twice)
        #pragma unroll
        for (int kc = 0; kc < GBK; kc += 32) {
            v16h a0 = load_a_frag(&As[cur][wr * 32 +      rr][kc], half);
            v16h a1 = load_a_frag(&As[cur][wr * 32 + 16 + rr][kc], half);
            v16h b0 = load_b_frag(&Bs[cur][wc * 32 +      rr][kc], half);
            v16h b1 = load_b_frag(&Bs[cur][wc * 32 + 16 + rr][kc], half);
            acc00 = wmma_f16(a0, b0, acc00);
            acc01 = wmma_f16(a0, b1, acc01);
            acc10 = wmma_f16(a1, b0, acc10);
            acc11 = wmma_f16(a1, b1, acc11);
        }

        // drain prefetched regs into the other buffer
        if (more) {
            const int nxt = cur ^ 1;
            #pragma unroll
            for (int i = 0; i < 4; ++i) *(v8h*)&As[nxt][ar][ac + i * 8] = areg[i];
            #pragma unroll
            for (int i = 0; i < 2; ++i) *(v8h*)&Bs[nxt][br][bc + i * 8] = breg[i];
            __syncthreads();
        }
    }

    // epilogue: C layout per tile: VGPR i -> row base+half*8+i, col base+rr
    #pragma unroll
    for (int ti = 0; ti < 2; ++ti) {
        #pragma unroll
        for (int tj = 0; tj < 2; ++tj) {
            v8f acc = (ti == 0) ? (tj == 0 ? acc00 : acc01)
                                : (tj == 0 ? acc10 : acc11);
            const int mbase = row0 + wr * 32 + ti * 16 + half * 8;
            const int n     = col0 + wc * 32 + tj * 16 + rr;
            if (OUT_F16) {
                _Float16* Cp = (_Float16*)C;
                #pragma unroll
                for (int i = 0; i < 8; ++i)
                    Cp[(size_t)(mbase + i) * ldc + n] = (_Float16)acc[i];
            } else {
                float* Cp = (float*)C;
                #pragma unroll
                for (int i = 0; i < 8; ++i)
                    Cp[(size_t)(mbase + i) * ldc + n] = acc[i];
            }
        }
    }
}

// ---------------------------------------------------------------------------
// Pass 3: in-place row softmax of S (f16), N columns, one block per row.
// ---------------------------------------------------------------------------
__global__ __launch_bounds__(256)
void softmax_rows_kernel(_Float16* __restrict__ S, int N) {
    __shared__ float red[256];
    const int t = threadIdx.x;
    const size_t base = (size_t)blockIdx.x * N;
    const int per = N / 256;              // 16 for N=4096

    float local[16];
    float m = -3.0e38f;
    #pragma unroll
    for (int i = 0; i < 16; ++i) {
        float v = (i < per) ? (float)S[base + t + i * 256] : -3.0e38f;
        local[i] = v;
        m = fmaxf(m, v);
    }
    red[t] = m; __syncthreads();
    for (int s = 128; s > 0; s >>= 1) {
        if (t < s) red[t] = fmaxf(red[t], red[t + s]);
        __syncthreads();
    }
    m = red[0]; __syncthreads();

    float sum = 0.0f;
    #pragma unroll
    for (int i = 0; i < 16; ++i) {
        if (i < per) {
            float e = __expf(local[i] - m);
            local[i] = e;
            sum += e;
        }
    }
    red[t] = sum; __syncthreads();
    for (int s = 128; s > 0; s >>= 1) {
        if (t < s) red[t] += red[t + s];
        __syncthreads();
    }
    const float inv = 1.0f / red[0];

    #pragma unroll
    for (int i = 0; i < 16; ++i)
        if (i < per) S[base + t + i * 256] = (_Float16)(local[i] * inv);
}

// ---------------------------------------------------------------------------
extern "C" void kernel_launch(void* const* d_in, const int* in_sizes, int n_in,
                              void* d_out, int out_size, void* d_ws, size_t ws_size,
                              hipStream_t stream) {
    (void)in_sizes; (void)n_in; (void)out_size; (void)ws_size;

    const float* x  = (const float*)d_in[0];
    const float* Wq = (const float*)d_in[1];
    const float* Wk = (const float*)d_in[2];
    const float* Wv = (const float*)d_in[3];
    float* out = (float*)d_out;

    // workspace layout (f16): Q 8MB | K 8MB | Vt 8MB | S/P 32MB   (56MB total)
    char* ws = (char*)d_ws;
    _Float16* Qh  = (_Float16*)(ws);
    _Float16* Kh  = (_Float16*)(ws + (8ull << 20));
    _Float16* Vth = (_Float16*)(ws + (16ull << 20));
    _Float16* Sh  = (_Float16*)(ws + (24ull << 20));

    const float kscale = 0.03125f;   // 1/sqrt(1024), folded into Q

    dim3 blk(256);
    dim3 gProj(SEQ / PBM, DMODEL / PBN);    // 64 x 16
    qkv_project_kernel<<<gProj, blk, 0, stream>>>(x, Wq, Qh,  kscale, 0);
    qkv_project_kernel<<<gProj, blk, 0, stream>>>(x, Wk, Kh,  1.0f,   0);
    qkv_project_kernel<<<gProj, blk, 0, stream>>>(x, Wv, Vth, 1.0f,   1);

    // S = Q K^T : M=N=SEQ, K=DMODEL (f16 out, stays in 192MB L2)
    dim3 gS(SEQ / GBM, SEQ / GBN);          // 32 x 64
    gemm_nt_kernel<1><<<gS, blk, 0, stream>>>(Qh, Kh, (void*)Sh,
                                              DMODEL, DMODEL, DMODEL, SEQ);

    // row softmax in place
    softmax_rows_kernel<<<dim3(SEQ), blk, 0, stream>>>(Sh, SEQ);

    // out = P Vt^T : M=SEQ, N=DMODEL, K=SEQ (f32 out)
    dim3 gO(SEQ / GBM, DMODEL / GBN);       // 32 x 16
    gemm_nt_kernel<0><<<gO, blk, 0, stream>>>(Sh, Vth, (void*)out,
                                              SEQ, SEQ, SEQ, DMODEL);
}
